// CausalSelfAttention_43181601194551
// MI455X (gfx1250) — matrix operand order repaired and checked by reading, not compiled, not run on hardware
//
#include <hip/hip_runtime.h>
#include <hip/hip_bf16.h>

// Problem constants (match reference)
#define B_    2
#define T_    2048
#define C_    1024
#define NH_   16
#define NKV_  4
#define HD_   64
#define NREP_ (NH_ / NKV_)
#define QKVD_ (NH_ * HD_ + 2 * NKV_ * HD_)   // 1536
#define QT_   (T_ / 16)
#define ROPE_BASE_ 10000.0f
#define EPS_ 1e-6f

typedef __attribute__((ext_vector_type(16))) __bf16        v16bf;
typedef __attribute__((ext_vector_type(8)))  float         v8f;
typedef __attribute__((ext_vector_type(4)))  unsigned int  u32x4;
using bf16_t = __hip_bfloat16;

union FragBF { v16bf v; u32x4 q[2]; };

__device__ __forceinline__ u32x4 ld128(const bf16_t* p) {
  return *(const u32x4*)p;
}

// CDNA5 async copy: global -> LDS, 16 bytes per lane, tracked by ASYNCcnt.
// LDS address is passed in a VGPR (low 32 bits of a generic __shared__ pointer).
__device__ __forceinline__ void async_ld_b128(bf16_t* lds_ptr,
                                              const bf16_t* gptr) {
  unsigned lds_off = (unsigned)(size_t)(void*)lds_ptr;
  asm volatile("global_load_async_to_lds_b128 %0, %1, off"
               :: "v"(lds_off), "v"(gptr)
               : "memory");
}
__device__ __forceinline__ void wait_async0() {
  asm volatile("s_wait_asynccnt 0x0" ::: "memory");
}

// ---------------------------------------------------------------------------
// fp32 -> bf16 convert (straight)
// ---------------------------------------------------------------------------
__global__ void cvt_bf16_kernel(const float* __restrict__ in,
                                bf16_t* __restrict__ out, int n) {
  int i = blockIdx.x * blockDim.x + threadIdx.x;
  if (i < n) out[i] = __float2bfloat16(in[i]);
}

// fp32 (rows x cols, row-major) -> bf16 transposed (cols x rows, row-major)
__global__ void transpose_cvt_kernel(const float* __restrict__ in,
                                     bf16_t* __restrict__ out,
                                     int rows, int cols) {
  int i = blockIdx.x * blockDim.x + threadIdx.x;
  if (i < rows * cols) {
    int k = i / cols;
    int n = i - k * cols;
    out[(size_t)n * rows + k] = __float2bfloat16(in[i]);
  }
}

// ---------------------------------------------------------------------------
// WMMA GEMM with async-LDS double buffering:
//   C[M x N] (fp32) = A[M x K] (bf16 row-major) * BT[N x K]^T (bf16)
// Block = 128 threads (4 waves); block tile 64x64; pipeline stage Kc = 64.
// A/B 64x64 tiles (8 KB each) staged in LDS via global_load_async_to_lds_b128,
// double buffered (32 KB LDS total): one s_wait_asynccnt + one split-barrier
// per stage amortized over 8 WMMAs and 8 async b128 copies per thread.
// ---------------------------------------------------------------------------
__global__ __launch_bounds__(128) void gemm_bf16f32_kernel(
    const bf16_t* __restrict__ A, const bf16_t* __restrict__ BT,
    float* __restrict__ Cout, int N, int K) {
  __shared__ __align__(16) bf16_t As[2][64 * 64];
  __shared__ __align__(16) bf16_t Bs[2][64 * 64];

  const int tid  = threadIdx.x;
  const int lane = tid & 31;
  const int wave = tid >> 5;
  const int half = lane >> 4;
  const int ln   = lane & 15;
  const int rowbase = blockIdx.y * 64;
  const int colbase = blockIdx.x * 64;

  // Each thread copies 4 x b128 per 64x64 tile (tile = 4096 bf16 = 512 b128).
  int e[4];
  const bf16_t* ap[4];
  const bf16_t* bp[4];
#pragma unroll
  for (int i = 0; i < 4; ++i) {
    e[i] = tid * 8 + i * 1024;
    const int ar = e[i] >> 6;      // tile row
    const int ac = e[i] & 63;      // tile col (K within stage)
    ap[i] = A  + (size_t)(rowbase + ar) * K + ac;
    bp[i] = BT + (size_t)(colbase + ar) * K + ac;
  }

  auto issue_copies = [&](int buf, int k0) {
#pragma unroll
    for (int i = 0; i < 4; ++i) async_ld_b128(&As[buf][e[i]], ap[i] + k0);
#pragma unroll
    for (int i = 0; i < 4; ++i) async_ld_b128(&Bs[buf][e[i]], bp[i] + k0);
  };

  issue_copies(0, 0);
  wait_async0();
  __syncthreads();

  v8f acc[4] = {};
  const int nk = K / 64;
  for (int kc = 0; kc < nk; ++kc) {
    const int cur = kc & 1;
    if (kc + 1 < nk) issue_copies(cur ^ 1, (kc + 1) * 64);

    // Two K=32 WMMA steps per stage (8 WMMAs), fragments from LDS.
#pragma unroll
    for (int kk = 0; kk < 64; kk += 32) {
      const bf16_t* arow = &As[cur][(wave * 16 + ln) * 64 + kk];
      FragBF a;
      a.q[0] = ld128(arow + half * 16);       // K = kk+half*16 .. +7
      a.q[1] = ld128(arow + half * 16 + 8);   // K = kk+half*16+8..+15
      FragBF bfr[4];
#pragma unroll
      for (int c = 0; c < 4; ++c) {
        const bf16_t* brow = &Bs[cur][(c * 16 + ln) * 64 + kk];
        bfr[c].q[0] = ld128(brow + half * 16);       // K = kk+half*16 .. +7
        bfr[c].q[1] = ld128(brow + half * 16 + 8);   // K = kk+half*16+8..+15
      }
#pragma unroll
      for (int c = 0; c < 4; ++c)
        acc[c] = __builtin_amdgcn_wmma_f32_16x16x32_bf16(
            false, a.v, false, bfr[c].v, (short)0, acc[c], false, false);
    }

    wait_async0();     // next-stage copies have landed (overlapped with WMMAs)
    __syncthreads();   // all waves done reading 'cur' before it is reused
  }

  // D layout: VGPR r -> (M = half*8 + r, N = ln)
#pragma unroll
  for (int c = 0; c < 4; ++c)
#pragma unroll
    for (int r = 0; r < 8; ++r)
      Cout[(size_t)(rowbase + wave * 16 + half * 8 + r) * N +
           colbase + c * 16 + ln] = acc[c][r];
}

// ---------------------------------------------------------------------------
// RoPE + per-head RMSNorm (fp32 math), emitting bf16 operands for attention:
//   qb [b][h][t][hd]   (softmax scale 1/sqrt(HD) folded in)
//   kb [b][g][t][hd]
//   vt [b][g][hd][t]   (transposed so PV B-fragments are contiguous)
// ---------------------------------------------------------------------------
__global__ void rope_rms_kernel(const float* __restrict__ qkv,
                                bf16_t* __restrict__ qb,
                                bf16_t* __restrict__ kb,
                                bf16_t* __restrict__ vt) {
  const int hTot = NH_ + 2 * NKV_;
  int tid = blockIdx.x * blockDim.x + threadIdx.x;
  if (tid >= B_ * T_ * hTot) return;
  const int h = tid % hTot;
  const int t = (tid / hTot) % T_;
  const int b = tid / (hTot * T_);

  const float* src = qkv + ((size_t)(b * T_ + t)) * QKVD_ + h * HD_;

  if (h < NH_ + NKV_) {
    float vr[HD_];
    float ss = 0.f;
#pragma unroll
    for (int i = 0; i < HD_ / 2; ++i) {
      float inv_freq = __powf(ROPE_BASE_, -(float)i / (float)(HD_ / 2));
      float freq = (float)t * inv_freq;
      float cs = cosf(freq), sn = sinf(freq);
      float a = src[2 * i], bv = src[2 * i + 1];
      float r0 = a * cs - bv * sn;
      float r1 = bv * cs + a * sn;
      vr[2 * i] = r0;
      vr[2 * i + 1] = r1;
      ss += r0 * r0 + r1 * r1;
    }
    float inv = rsqrtf(ss * (1.0f / HD_) + EPS_);
    if (h < NH_) {
      inv *= 0.125f;  // fold 1/sqrt(HD) softmax scale into Q
      bf16_t* dst = qb + (((size_t)(b * NH_ + h)) * T_ + t) * HD_;
#pragma unroll
      for (int i = 0; i < HD_; ++i) dst[i] = __float2bfloat16(vr[i] * inv);
    } else {
      const int g = h - NH_;
      bf16_t* dst = kb + (((size_t)(b * NKV_ + g)) * T_ + t) * HD_;
#pragma unroll
      for (int i = 0; i < HD_; ++i) dst[i] = __float2bfloat16(vr[i] * inv);
    }
  } else {
    const int g = h - NH_ - NKV_;
    bf16_t* dst = vt + ((size_t)(b * NKV_ + g)) * HD_ * T_;
#pragma unroll
    for (int i = 0; i < HD_; ++i)
      dst[(size_t)i * T_ + t] = __float2bfloat16(src[i]);
  }
}

// ---------------------------------------------------------------------------
// Flash attention: one wave (32 threads) per (b, h, 16-row Q tile).
// Software-pipelined: K fragments for block j+1 are loaded immediately after
// the S-WMMAs of block j consume them, and V fragments are issued before the
// softmax, so the fp32 softmax VALU section hides all global-load latency.
// Tile-1 K loads use a clamped offset on the final ragged block (results are
// fully causal-masked), keeping the loop branch-free and EXEC uniform.
// ---------------------------------------------------------------------------
__global__ __launch_bounds__(32) void attn_kernel(
    const bf16_t* __restrict__ qb, const bf16_t* __restrict__ kb,
    const bf16_t* __restrict__ vt, bf16_t* __restrict__ yb) {
  __shared__ __align__(16) bf16_t Pl[16 * 32];

  const int lane = threadIdx.x & 31;
  const int half = lane >> 4;
  const int ln   = lane & 15;
  int idx = blockIdx.x;
  const int qt = idx % QT_; idx /= QT_;
  const int h  = idx % NH_;
  const int b  = idx / NH_;
  const int g  = h / NREP_;
  const int t0 = qt * 16;
  const int kend = t0 + 16;

  // Q A-fragments for both HD chunks ([0,32), [32,64)); row M = ln
  const bf16_t* qrow = qb + (((size_t)(b * NH_ + h)) * T_ + t0 + ln) * HD_;
  FragBF qf0, qf1;
  qf0.q[0] = ld128(qrow + half * 16);
  qf0.q[1] = ld128(qrow + half * 16 + 8);
  qf1.q[0] = ld128(qrow + 32 + half * 16);
  qf1.q[1] = ld128(qrow + 32 + half * 16 + 8);

  const bf16_t* kB = kb + ((size_t)(b * NKV_ + g)) * T_ * HD_;
  const bf16_t* vB = vt + ((size_t)(b * NKV_ + g)) * HD_ * T_;

  // K fragments: (kfa,kfb) = 16-col tile 0 (hd chunks 0/1),
  //              (kfc,kfd) = 16-col tile 1 (clamped on ragged last block).
  FragBF kfa, kfb, kfc, kfd;
  auto load_k = [&](int k0) {
    const int off2 = (k0 + 16 < kend) ? 16 : 0;
    const bf16_t* kr0 = kB + (size_t)(k0 + ln) * HD_;
    const bf16_t* kr1 = kB + (size_t)(k0 + off2 + ln) * HD_;
    kfa.q[0] = ld128(kr0 + half * 16);
    kfa.q[1] = ld128(kr0 + half * 16 + 8);
    kfb.q[0] = ld128(kr0 + 32 + half * 16);
    kfb.q[1] = ld128(kr0 + 32 + half * 16 + 8);
    kfc.q[0] = ld128(kr1 + half * 16);
    kfc.q[1] = ld128(kr1 + half * 16 + 8);
    kfd.q[0] = ld128(kr1 + 32 + half * 16);
    kfd.q[1] = ld128(kr1 + 32 + half * 16 + 8);
  };
  load_k(0);

  v8f o[4] = {};
  float m[8], l[8];
#pragma unroll
  for (int r = 0; r < 8; ++r) { m[r] = -1e30f; l[r] = 0.f; }

  for (int kb0 = 0; kb0 < kend; kb0 += 32) {
    const bool has2 = (kb0 + 16) < kend;

    // ---- S = Q * K^T (4 WMMAs; tile 1 may be garbage -> masked below) ----
    v8f s0 = {}, s1 = {};
    s0 = __builtin_amdgcn_wmma_f32_16x16x32_bf16(false, qf0.v, false, kfa.v,
                                                 (short)0, s0, false, false);
    s0 = __builtin_amdgcn_wmma_f32_16x16x32_bf16(false, qf1.v, false, kfb.v,
                                                 (short)0, s0, false, false);
    s1 = __builtin_amdgcn_wmma_f32_16x16x32_bf16(false, qf0.v, false, kfc.v,
                                                 (short)0, s1, false, false);
    s1 = __builtin_amdgcn_wmma_f32_16x16x32_bf16(false, qf1.v, false, kfd.v,
                                                 (short)0, s1, false, false);

    // ---- prefetch next block's K fragments (hidden under softmax) ----
    if (kb0 + 32 < kend) load_k(kb0 + 32);

    // ---- issue V fragment loads now (also hidden under softmax) ----
    const int koff2 = has2 ? 16 : 0;  // clamp (P==0 there anyway)
    FragBF vf[4];
#pragma unroll
    for (int c = 0; c < 4; ++c) {
      const bf16_t* vcol = vB + (size_t)(c * 16 + ln) * T_ + kb0;
      vf[c].q[0] = ld128(vcol + half * koff2);     // K rows half*16 .. +7
      vf[c].q[1] = ld128(vcol + half * koff2 + 8);
    }

    // ---- online softmax (causal mask, per-row stats) ----
#pragma unroll
    for (int r = 0; r < 8; ++r) {
      const int row = t0 + half * 8 + r;
      const int c0 = kb0 + ln;
      const int c1 = kb0 + 16 + ln;
      float a0 = (c0 <= row) ? s0[r] : -1e30f;
      float a1 = (has2 && c1 <= row) ? s1[r] : -1e30f;
      float mx = fmaxf(a0, a1);
      mx = fmaxf(mx, __shfl_xor(mx, 1));
      mx = fmaxf(mx, __shfl_xor(mx, 2));
      mx = fmaxf(mx, __shfl_xor(mx, 4));
      mx = fmaxf(mx, __shfl_xor(mx, 8));
      float mnew = fmaxf(m[r], mx);
      float sc = __expf(m[r] - mnew);
      float p0 = __expf(a0 - mnew);
      float p1 = __expf(a1 - mnew);
      float rs = p0 + p1;
      rs += __shfl_xor(rs, 1);
      rs += __shfl_xor(rs, 2);
      rs += __shfl_xor(rs, 4);
      rs += __shfl_xor(rs, 8);
      l[r] = l[r] * sc + rs;
      m[r] = mnew;
      o[0][r] *= sc; o[1][r] *= sc; o[2][r] *= sc; o[3][r] *= sc;
      const int prow = half * 8 + r;
      Pl[prow * 32 + ln]      = __float2bfloat16(p0);
      Pl[prow * 32 + 16 + ln] = __float2bfloat16(p1);
    }
    asm volatile("s_wait_dscnt 0x0" ::: "memory");  // LDS stores visible
    __builtin_amdgcn_wave_barrier();

    // ---- P (16x32, A-frag from LDS) * V (32x64) ----
    FragBF pf;
    const bf16_t* pr = Pl + (size_t)ln * 32;        // row M = ln
    pf.q[0] = ld128(pr + half * 16);                // K = half*16 .. +7
    pf.q[1] = ld128(pr + half * 16 + 8);            // K = half*16 + 8 .. +15
#pragma unroll
    for (int c = 0; c < 4; ++c)
      o[c] = __builtin_amdgcn_wmma_f32_16x16x32_bf16(
          false, pf.v, false, vf[c].v, (short)0, o[c], false, false);
  }

  // ---- normalize and store y (bf16, [b][t][h*HD+col]) ----
#pragma unroll
  for (int r = 0; r < 8; ++r) {
    const float inv = 1.0f / l[r];
    const int row = t0 + half * 8 + r;
    bf16_t* dst = yb + ((size_t)(b * T_ + row)) * C_ + h * HD_;
#pragma unroll
    for (int c = 0; c < 4; ++c)
      dst[c * 16 + ln] = __float2bfloat16(o[c][r] * inv);
  }
}

// ---------------------------------------------------------------------------
// Host-side launch
// ---------------------------------------------------------------------------
extern "C" void kernel_launch(void* const* d_in, const int* in_sizes, int n_in,
                              void* d_out, int out_size, void* d_ws,
                              size_t ws_size, hipStream_t stream) {
  const float* x      = (const float*)d_in[0];
  const float* w_attn = (const float*)d_in[1];
  const float* w_proj = (const float*)d_in[2];
  float* out = (float*)d_out;

  char* p = (char*)d_ws;
  auto take = [&](size_t bytes) -> char* {
    char* r = p;
    p += (bytes + 255) & ~(size_t)255;
    return r;
  };
  bf16_t* xb  = (bf16_t*)take((size_t)B_ * T_ * C_ * 2);
  bf16_t* waT = (bf16_t*)take((size_t)QKVD_ * C_ * 2);
  bf16_t* wpT = (bf16_t*)take((size_t)C_ * C_ * 2);
  float*  qkv = (float*) take((size_t)B_ * T_ * QKVD_ * 4);
  bf16_t* qb  = (bf16_t*)take((size_t)B_ * NH_ * T_ * HD_ * 2);
  bf16_t* kb  = (bf16_t*)take((size_t)B_ * NKV_ * T_ * HD_ * 2);
  bf16_t* vt  = (bf16_t*)take((size_t)B_ * NKV_ * HD_ * T_ * 2);
  bf16_t* yb  = (bf16_t*)take((size_t)B_ * T_ * C_ * 2);

  // 1) precision conversion / weight transposes
  const int nx = B_ * T_ * C_;
  cvt_bf16_kernel<<<(nx + 255) / 256, 256, 0, stream>>>(x, xb, nx);
  transpose_cvt_kernel<<<(C_ * QKVD_ + 255) / 256, 256, 0, stream>>>(
      w_attn, waT, C_, QKVD_);
  transpose_cvt_kernel<<<(C_ * C_ + 255) / 256, 256, 0, stream>>>(
      w_proj, wpT, C_, C_);

  // 2) QKV projection (4096 x 1536 = 4096 x 1024 @ 1024 x 1536)
  dim3 g1(QKVD_ / 64, (B_ * T_) / 64);
  gemm_bf16f32_kernel<<<g1, 128, 0, stream>>>(xb, waT, qkv, QKVD_, C_);

  // 3) RoPE + RMSNorm + layout for attention
  const int nr = B_ * T_ * (NH_ + 2 * NKV_);
  rope_rms_kernel<<<(nr + 255) / 256, 256, 0, stream>>>(qkv, qb, kb, vt);

  // 4) causal flash attention (one wave per 16-row Q tile)
  attn_kernel<<<B_ * NH_ * QT_, 32, 0, stream>>>(qb, kb, vt, yb);

  // 5) output projection (4096 x 1024 @ 1024 x 1024) -> fp32 out
  dim3 g2(C_ / 64, (B_ * T_) / 64);
  gemm_bf16f32_kernel<<<g2, 128, 0, stream>>>(yb, wpT, out, C_, C_);
}
